// RnnSearchModule_62259845923321
// MI455X (gfx1250) — matrix operand order repaired and checked by reading, not compiled, not run on hardware
//
#include <hip/hip_runtime.h>
#include <cmath>

// ---------------------------------------------------------------------------
// RNN (tanh cell + batch-norm state) for MI455X / gfx1250, wave32 WMMA bf16.
//   T=512, B=64, D_IN=D_H=1024.
//   prep_kernel : W_x, W_h (fp32 [K][N]) -> bf16 transposed [N][K] in ws
//   xproj_kernel: Xp = X @ W_x + b, double-buffered LDS, TDM weight staging
//   rnn_kernel  : persistent, 16 WGs x 128 thr, 512 steps; TDM double-buffered
//                 LDS staging; device-scope spin barrier; WG-local BN.
// ---------------------------------------------------------------------------

typedef __attribute__((ext_vector_type(16))) __bf16   v16bf;
typedef __attribute__((ext_vector_type(8)))  float    v8f;
typedef __attribute__((ext_vector_type(4)))  unsigned u32x4;
typedef __attribute__((ext_vector_type(2)))  unsigned u32x2;
typedef __attribute__((ext_vector_type(4)))  float    f32x4;
typedef __attribute__((ext_vector_type(4)))  unsigned tdmU4;
typedef __attribute__((ext_vector_type(8)))  int      tdmI8;
typedef __attribute__((ext_vector_type(4)))  int      tdmI4;

#define T_STEPS 512
#define B_DIM   64
#define D_DIM   1024
#define N_WGS   16
#define EPS     1e-5f

#if __has_builtin(__builtin_amdgcn_tensor_load_to_lds) && \
    __has_builtin(__builtin_amdgcn_s_wait_tensorcnt)
#define USE_TDM 1
#pragma message("CDNA5 marker: TDM path ENABLED (tensor_load_to_lds)")
#else
#define USE_TDM 0
#pragma message("CDNA5 marker: TDM builtin NOT available, vector-staging fallback")
#endif
#define XSTR2(x) #x
#define XSTR(x) XSTR2(x)
#pragma message("CDNA5 marker: clang major = " XSTR(__clang_major__))

union FragBF { u32x4 u[2]; v16bf v; };
union Pack4  { __bf16 h[4]; u32x2 u; };

// --- weight transpose + fp32 -> bf16 ---------------------------------------
__global__ __launch_bounds__(256) void prep_kernel(const float* __restrict__ Wx,
                                                   const float* __restrict__ Wh,
                                                   __bf16* __restrict__ WXT,
                                                   __bf16* __restrict__ WHT) {
    const float* src = blockIdx.y ? Wh : Wx;
    __bf16*      dst = blockIdx.y ? WHT : WXT;
    int idx = blockIdx.x * 256 + threadIdx.x;      // over 1024*1024
    int n = idx >> 10, k = idx & 1023;
    dst[idx] = (__bf16)src[k * D_DIM + n];         // dst[n][k] = src[k][n]
}

// --- 2x2 register-tile WMMA step over one 64-K chunk ------------------------
__device__ __forceinline__ v8f wmma_bf16(const v16bf& a, const v16bf& b, v8f c) {
    return __builtin_amdgcn_wmma_f32_16x16x32_bf16(false, a, false, b,
                                                   (short)0, c, false, false);
}

__device__ __forceinline__ void wmma_step22(const __bf16* As, const __bf16* Bs,
                                            int mi0, int ni0, int g, int lm,
                                            v8f& c00, v8f& c01, v8f& c10, v8f& c11) {
    const int rA0 = (mi0 * 16 + lm) * 64;
    const int rA1 = rA0 + 16 * 64;
    const int rB0 = (ni0 * 16 + lm) * 64;
    const int rB1 = rB0 + 16 * 64;
#pragma unroll
    for (int kk = 0; kk < 64; kk += 32) {
        FragBF a0, a1, b0, b1;
        // A 16x32 bf16: lane group g, elems 0..7 -> K=8g.., 8..15 -> K=16+8g..
        a0.u[0] = *(const u32x4*)&As[rA0 + kk + 8 * g];
        a0.u[1] = *(const u32x4*)&As[rA0 + kk + 16 + 8 * g];
        a1.u[0] = *(const u32x4*)&As[rA1 + kk + 8 * g];
        a1.u[1] = *(const u32x4*)&As[rA1 + kk + 16 + 8 * g];
        // B 32x16 bf16 (rows of W^T): lane group g, elems 0..7 -> K=8g.., 8..15 -> K=16+8g..
        b0.u[0] = *(const u32x4*)&Bs[rB0 + kk + 8 * g];
        b0.u[1] = *(const u32x4*)&Bs[rB0 + kk + 16 + 8 * g];
        b1.u[0] = *(const u32x4*)&Bs[rB1 + kk + 8 * g];
        b1.u[1] = *(const u32x4*)&Bs[rB1 + kk + 16 + 8 * g];
        c00 = wmma_bf16(a0.v, b0.v, c00);
        c01 = wmma_bf16(a0.v, b1.v, c01);
        c10 = wmma_bf16(a1.v, b0.v, c10);
        c11 = wmma_bf16(a1.v, b1.v, c11);
    }
}

// --- two-phase bf16 tile staging (batch loads, then stores) -----------------
template <int NTHR, int NROWS>
__device__ __forceinline__ void stage_bf16_rows(__bf16* dst, const __bf16* src,
                                                int row0, int pitch, int kc, int tid) {
    constexpr int ITERS = (NROWS * 8) / NTHR;
    u32x4 tmp[ITERS];
#pragma unroll
    for (int j = 0; j < ITERS; ++j) {
        int i = tid + j * NTHR;
        int r = i >> 3, c = (i & 7) * 8;
        tmp[j] = *(const u32x4*)&src[(size_t)(row0 + r) * pitch + kc + c];
    }
#pragma unroll
    for (int j = 0; j < ITERS; ++j) {
        int i = tid + j * NTHR;
        int r = i >> 3, c = (i & 7) * 8;
        *(u32x4*)&dst[r * 64 + c] = tmp[j];
    }
}

#if USE_TDM
// --- Tensor Data Mover: 2-D bf16 tile (rows x 64) global -> LDS -------------
// D# per CDNA5 ISA 8.3/8.4: group0 = {count, lds_addr, global_addr, type=2},
// group1 = {data_size=2B, tensor dims (large: tiles always in-bounds),
//           tile_dim0=cols, tile_dim1=rows, dim0 stride}.
__device__ __forceinline__ void tdm_load_tile_bf16(unsigned lds_addr,
                                                   const __bf16* gptr,
                                                   int rows, int cols,
                                                   unsigned long pitch_elems) {
    unsigned long ga = (unsigned long)(__UINTPTR_TYPE__)gptr;
    tdmU4 g0 = {};
    g0[0] = 1u;                                          // count=1, user mode
    g0[1] = lds_addr;                                    // LDS byte address
    g0[2] = (unsigned)ga;                                // global_addr[31:0]
    g0[3] = (unsigned)((ga >> 32) & 0x01FFFFFFu) | (2u << 30);  // [56:32], type=2
    const unsigned td0 = 0x40000000u, td1 = 0x40000000u; // huge dims: no OOB clamp
    tdmI8 g1 = {};
    g1[0] = (int)(1u << 16);                             // data_size=1 (2 bytes)
    g1[1] = (int)((td0 & 0xFFFFu) << 16);                // tensor_dim0[15:0]
    g1[2] = (int)((td0 >> 16) | ((td1 & 0xFFFFu) << 16));// dim0[31:16], dim1[15:0]
    g1[3] = (int)((td1 >> 16) | ((unsigned)cols << 16)); // dim1[31:16], tile_dim0
    g1[4] = rows;                                        // tile_dim1 (tile_dim2=0)
    g1[5] = (int)(unsigned)pitch_elems;                  // dim0_stride[31:0]
    g1[6] = (int)((unsigned)(pitch_elems >> 32) & 0xFFFFu); // dim0_stride[47:32]
    g1[7] = 0;
    tdmI4 g2 = {}, g3 = {};
#if defined(__clang_major__) && __clang_major__ >= 23
    tdmI8 g1b = {};
    __builtin_amdgcn_tensor_load_to_lds(g0, g1, g2, g3, g1b, 0);
#else
    __builtin_amdgcn_tensor_load_to_lds(g0, g1, g2, g3, 0);
#endif
}
#endif // USE_TDM

// --- phase 1: Xp = X @ W_x + b  -> out2 region (double-buffered pipeline) ---
__global__ __launch_bounds__(256) void xproj_kernel(const float*  __restrict__ X,
                                                    const __bf16* __restrict__ WXT,
                                                    const float*  __restrict__ bias,
                                                    float* __restrict__ Xp) {
    __shared__ __bf16 AsB[2][64 * 64];    // 64 M x 64 K, double-buffered
    __shared__ __bf16 BsB[2][128 * 64];   // 128 N x 64 K (W^T), double-buffered
    const int tid  = threadIdx.x;
    const int wave = tid >> 5, lane = tid & 31;
    const int g = lane >> 4, lm = lane & 15;
    const int mi0 = (wave & 1) * 2;            // 2x2 tile block per wave
    const int ni0 = (wave >> 1) * 2;           // 8 waves cover 4 x 8 tiles
    const int m0 = blockIdx.x * 64;            // row tile (T*B rows)
    const int n0 = blockIdx.y * 128;           // column tile

    // A staging helpers: thread owns 4 fixed float4 slots of the 64x64 tile
    f32x4 fbuf[4];
    auto loadA = [&](int kc) {
#pragma unroll
        for (int j = 0; j < 4; ++j) {
            int i = tid + j * 256;
            int r = i >> 4, c = (i & 15) * 4;
            fbuf[j] = *(const f32x4*)&X[(size_t)(m0 + r) * D_DIM + kc + c];
        }
    };
    auto storeA = [&](__bf16* dst) {
#pragma unroll
        for (int j = 0; j < 4; ++j) {
            int i = tid + j * 256;
            int r = i >> 4, c = (i & 15) * 4;
            Pack4 p;
            p.h[0] = (__bf16)fbuf[j].x; p.h[1] = (__bf16)fbuf[j].y;
            p.h[2] = (__bf16)fbuf[j].z; p.h[3] = (__bf16)fbuf[j].w;
            *(u32x2*)&dst[r * 64 + c] = p.u;
        }
    };

    // prologue: chunk 0 into buffer 0
    loadA(0);
    storeA(AsB[0]);
#if USE_TDM
    if (wave == 0) {
        tdm_load_tile_bf16((unsigned)(__UINTPTR_TYPE__)&BsB[0][0],
                           WXT + (size_t)n0 * D_DIM, 128, 64, D_DIM);
        __builtin_amdgcn_s_wait_tensorcnt(0);
    }
#else
    stage_bf16_rows<256, 128>(BsB[0], WXT, n0, D_DIM, 0, tid);
#endif
    __syncthreads();

    v8f c00 = {}, c01 = {}, c10 = {}, c11 = {};
#pragma unroll 1
    for (int kc = 0, ib = 0; kc < D_DIM; kc += 64, ib ^= 1) {
        const bool more = (kc + 64) < D_DIM;
        if (more) {
            loadA(kc + 64);                       // global loads issued before wmma
#if USE_TDM
            if (wave == 0)
                tdm_load_tile_bf16((unsigned)(__UINTPTR_TYPE__)&BsB[ib ^ 1][0],
                                   WXT + (size_t)n0 * D_DIM + (kc + 64),
                                   128, 64, D_DIM);
#endif
        }
        wmma_step22(AsB[ib], BsB[ib], mi0, ni0, g, lm, c00, c01, c10, c11);
        if (more) {
            storeA(AsB[ib ^ 1]);
#if USE_TDM
            if (wave == 0) __builtin_amdgcn_s_wait_tensorcnt(0);
#else
            stage_bf16_rows<256, 128>(BsB[ib ^ 1], WXT, n0, D_DIM, kc + 64, tid);
#endif
        }
        __syncthreads();
    }

    // epilogue: + bias. C layout: vgpr r -> M = 8g + r, N = lane%16
    const int Mg0 = m0 + mi0 * 16 + 8 * g;
    const int col0 = n0 + ni0 * 16 + lm;
    const float bv0 = bias[col0], bv1 = bias[col0 + 16];
#pragma unroll
    for (int r = 0; r < 8; ++r) {
        Xp[(size_t)(Mg0 + r) * D_DIM + col0]           = c00[r] + bv0;
        Xp[(size_t)(Mg0 + r) * D_DIM + col0 + 16]      = c01[r] + bv1;
        Xp[(size_t)(Mg0 + 16 + r) * D_DIM + col0]      = c10[r] + bv0;
        Xp[(size_t)(Mg0 + 16 + r) * D_DIM + col0 + 16] = c11[r] + bv1;
    }
}

// --- device-scope spin barrier (16 resident WGs) ----------------------------
__device__ __forceinline__ void grid_barrier(unsigned* ctr, unsigned target) {
    __threadfence();
    __syncthreads();
    if (threadIdx.x == 0) {
        __hip_atomic_fetch_add(ctr, 1u, __ATOMIC_RELEASE, __HIP_MEMORY_SCOPE_AGENT);
        while (__hip_atomic_load(ctr, __ATOMIC_ACQUIRE, __HIP_MEMORY_SCOPE_AGENT) < target)
            __builtin_amdgcn_s_sleep(2);
    }
    __syncthreads();
}

// --- phase 2: persistent recurrence -----------------------------------------
__global__ __launch_bounds__(128) void rnn_kernel(const float*  __restrict__ state0,
                                                  const __bf16* __restrict__ WHT,
                                                  float* __restrict__ out1,
                                                  float* __restrict__ out2,  // holds Xp
                                                  __bf16* __restrict__ S,
                                                  unsigned* __restrict__ ctr) {
    __shared__ __bf16 AsB[2][64 * 64];    // double-buffered state chunk
    __shared__ __bf16 BsB[2][64 * 64];    // double-buffered W_h^T chunk
    __shared__ float  Hs[64 * 64];
    __shared__ float  muS[64], rsS[64];

    const int tid  = threadIdx.x;
    const int wave = tid >> 5, lane = tid & 31;
    const int g = lane >> 4, lm = lane & 15;
    const int mi0 = (wave & 1) * 2;            // 4 waves cover 4 x 4 tiles, 2x2 each
    const int ni0 = (wave >> 1) * 2;
    const int wg = blockIdx.x;                 // 0..15
    const int n0 = wg * 64;

    // init bf16 state slice from state_tensor[0]
    for (int i = tid; i < 64 * 64; i += 128) {
        int r = i >> 6, c = i & 63;
        S[(size_t)r * D_DIM + n0 + c] = (__bf16)state0[(size_t)r * D_DIM + n0 + c];
    }
    unsigned target = N_WGS;
    grid_barrier(ctr, target);

#pragma unroll 1
    for (int t = 0; t < T_STEPS; ++t) {
        v8f c00 = {}, c01 = {}, c10 = {}, c11 = {};

#if USE_TDM
        if (wave == 0) {   // chunk 0 -> buffer 0 (TDM, EXEC-independent)
            tdm_load_tile_bf16((unsigned)(__UINTPTR_TYPE__)&AsB[0][0], S, 64, 64, D_DIM);
            tdm_load_tile_bf16((unsigned)(__UINTPTR_TYPE__)&BsB[0][0],
                               WHT + (size_t)n0 * D_DIM, 64, 64, D_DIM);
        }
#endif
#pragma unroll 1
        for (int kc = 0, ib = 0; kc < D_DIM; kc += 64, ib ^= 1) {
            __syncthreads();   // everyone done reading buffer ib^1
#if USE_TDM
            if (wave == 0) {
                const bool more = (kc + 64) < D_DIM;
                if (more) {    // prefetch chunk kc+64 into the other buffer
                    tdm_load_tile_bf16((unsigned)(__UINTPTR_TYPE__)&AsB[ib ^ 1][0],
                                       S + (kc + 64), 64, 64, D_DIM);
                    tdm_load_tile_bf16((unsigned)(__UINTPTR_TYPE__)&BsB[ib ^ 1][0],
                                       WHT + (size_t)n0 * D_DIM + (kc + 64),
                                       64, 64, D_DIM);
                    __builtin_amdgcn_s_wait_tensorcnt(2);  // current chunk landed
                } else {
                    __builtin_amdgcn_s_wait_tensorcnt(0);
                }
            }
#else
            stage_bf16_rows<128, 64>(AsB[ib], S,   0,  D_DIM, kc, tid);
            stage_bf16_rows<128, 64>(BsB[ib], WHT, n0, D_DIM, kc, tid);
#endif
            __syncthreads();   // buffer ib ready
            wmma_step22(AsB[ib], BsB[ib], mi0, ni0, g, lm, c00, c01, c10, c11);
        }

        // h = tanh(acc + Xp[t]) into LDS (Hs[batch][local_col])
        const size_t tbase = (size_t)t * B_DIM * D_DIM;
        const int Ml0 = mi0 * 16 + 8 * g;
        const int cl0 = ni0 * 16 + lm;
#pragma unroll
        for (int r = 0; r < 8; ++r) {
            float x00 = out2[tbase + (size_t)(Ml0 + r) * D_DIM + n0 + cl0];
            float x01 = out2[tbase + (size_t)(Ml0 + r) * D_DIM + n0 + cl0 + 16];
            float x10 = out2[tbase + (size_t)(Ml0 + 16 + r) * D_DIM + n0 + cl0];
            float x11 = out2[tbase + (size_t)(Ml0 + 16 + r) * D_DIM + n0 + cl0 + 16];
            Hs[(Ml0 + r) * 64 + cl0]           = tanhf(c00[r] + x00);
            Hs[(Ml0 + r) * 64 + cl0 + 16]      = tanhf(c01[r] + x01);
            Hs[(Ml0 + 16 + r) * 64 + cl0]      = tanhf(c10[r] + x10);
            Hs[(Ml0 + 16 + r) * 64 + cl0 + 16] = tanhf(c11[r] + x11);
        }
        __syncthreads();

        // batch-norm stats over 64 batch rows (one column per thread, bank-clean)
        if (tid < 64) {
            float s = 0.f, s2 = 0.f;
#pragma unroll 4
            for (int r = 0; r < 64; ++r) {
                float v = Hs[r * 64 + tid];
                s += v; s2 += v * v;
            }
            float mu  = s * (1.0f / 64.0f);
            float var = s2 * (1.0f / 64.0f) - mu * mu;   // biased variance
            muS[tid] = mu;
            rsS[tid] = rsqrtf(var + EPS);
        }
        __syncthreads();

        // normalize; write both output copies + next bf16 state
        for (int i = tid; i < 64 * 64; i += 128) {
            int r = i >> 6, c = i & 63;
            float v = (Hs[r * 64 + c] - muS[c]) * rsS[c];
            size_t gi = tbase + (size_t)r * D_DIM + n0 + c;
            out1[gi] = v;
            out2[gi] = v;                                // overwrites consumed Xp tile
            S[(size_t)r * D_DIM + n0 + c] = (__bf16)v;
        }

        target += N_WGS;
        grid_barrier(ctr, target);
    }
}

// ---------------------------------------------------------------------------
extern "C" void kernel_launch(void* const* d_in, const int* in_sizes, int n_in,
                              void* d_out, int out_size, void* d_ws, size_t ws_size,
                              hipStream_t stream) {
    const float* X  = (const float*)d_in[0];   // [512,64,1024]
    const float* S0 = (const float*)d_in[1];   // [1,64,1024]
    const float* Wx = (const float*)d_in[2];   // [1024,1024]
    const float* Wh = (const float*)d_in[3];   // [1024,1024]
    const float* b  = (const float*)d_in[4];   // [1024]

    float* out1 = (float*)d_out;                                  // copy 1
    float* out2 = out1 + (size_t)T_STEPS * B_DIM * D_DIM;         // copy 2 (Xp scratch)

    // workspace (~4.3 MB): [ctr pad 256B][WXT 2MB][WHT 2MB][S 128KB]
    char* ws = (char*)d_ws;
    unsigned* ctr = (unsigned*)ws;
    __bf16* WXT = (__bf16*)(ws + 256);
    __bf16* WHT = WXT + (size_t)D_DIM * D_DIM;
    __bf16* S   = WHT + (size_t)D_DIM * D_DIM;

    hipMemsetAsync(ws, 0, 256, stream);   // reset barrier counter every call

    prep_kernel <<<dim3(4096, 2),                          256, 0, stream>>>(Wx, Wh, WXT, WHT);
    xproj_kernel<<<dim3(T_STEPS * B_DIM / 64, D_DIM / 128), 256, 0, stream>>>(X, WXT, b, out2);
    rnn_kernel  <<<N_WGS,                                  128, 0, stream>>>(S0, WHT, out1, out2, S, ctr);
}
